// LinearAttention3D_8392366096918
// MI455X (gfx1250) — matrix operand order repaired and checked by reading, not compiled, not run on hardware
//
#include <hip/hip_runtime.h>

// -------- problem constants (from reference) --------
#define NSP   32768      // N = D*H*W = 32*32*32
#define CCH   256        // C
#define O3    768        // 3*C
#define BB    2          // batch
#define NHEADS 8
#define HD    32         // head dim
#define EPSV  1e-5f
#define QKV_NT 16        // N-tiles per block pass
#define BPITCH 264       // LDS row pitch in ushorts (256 + 8 pad -> 4-bank row skew)

// -------- WMMA helper types (bf16 16x16x32, f32 acc) --------
typedef __attribute__((ext_vector_type(16))) __bf16         bf16x16;
typedef __attribute__((ext_vector_type(8)))  float          f32x8;
typedef __attribute__((ext_vector_type(8)))  unsigned short u16x8;

union TileU {
  bf16x16 bf;
  u16x8   h[2];
  unsigned short us[16];
};

__device__ __forceinline__ unsigned short f2bf(float f) {
  unsigned int u = __float_as_uint(f);
  u = (u + 0x7FFFu + ((u >> 16) & 1u)) >> 16;   // round-to-nearest-even
  return (unsigned short)u;
}
__device__ __forceinline__ float bf2f(unsigned short s) {
  return __uint_as_float(((unsigned int)s) << 16);
}

// A-tile: 16(M) x 32(K) bf16 from row-major [row][k], k contiguous (ISA 7.12.2).
__device__ __forceinline__ TileU load_a_tile(const unsigned short* base, int ld,
                                             int m0, int k0, int lane) {
  const int m = lane & 15, half = lane >> 4;
  const unsigned short* p = base + (size_t)(m0 + m) * ld + (k0 + half * 16);
  TileU t;
  t.h[0] = *(const u16x8*)(p);
  t.h[1] = *(const u16x8*)(p + 8);
  return t;
}

// B-tile: 32(K) x 16(N) bf16 from a K-contiguous layout src[n][k] (global).
__device__ __forceinline__ TileU load_b_tile(const unsigned short* base, int ld,
                                             int n0, int k0, int lane) {
  const int n = lane & 15, kg = lane >> 4;
  const unsigned short* p = base + (size_t)(n0 + n) * ld + (k0 + kg * 16);
  TileU t;
  t.h[0] = *(const u16x8*)(p);
  t.h[1] = *(const u16x8*)(p + 8);
  return t;
}

// B-tile from padded LDS tile [16 rows][BPITCH] -> ds_load_b128 pairs.
__device__ __forceinline__ TileU load_b_tile_lds(const unsigned short* buf, int k0, int lane) {
  const int n = lane & 15, kg = lane >> 4;
  const unsigned short* p = buf + n * BPITCH + k0 + kg * 16;
  TileU t;
  t.h[0] = *(const u16x8*)(p);
  t.h[1] = *(const u16x8*)(p + 8);
  return t;
}

__device__ __forceinline__ f32x8 wmma_bf16(const TileU& a, const TileU& b, f32x8 c) {
  return __builtin_amdgcn_wmma_f32_16x16x32_bf16(false, a.bf, false, b.bf,
                                                 (short)0, c, false, false);
}

// ---- CDNA5 async global->LDS copy (ASYNCcnt path, ISA 10.x / 15.18.3) ----
__device__ __forceinline__ unsigned lds_off32(const void* p) {
  // flat LDS-aperture address: addr[31:0] == LDS byte offset
  return (unsigned)(unsigned long long)(uintptr_t)p;
}
__device__ __forceinline__ void async_copy16(const unsigned short* gptr, unsigned ldsoff) {
  unsigned long long ga = (unsigned long long)(uintptr_t)gptr;
  asm volatile("global_load_async_to_lds_b128 %0, %1, off"
               :: "v"(ldsoff), "v"(ga) : "memory");
}
__device__ __forceinline__ void wait_async0() {
  asm volatile("s_wait_asynccnt 0" ::: "memory");
}

// ============ f32 -> bf16 weight convert ============
__global__ __launch_bounds__(256) void f2bf_convert_kernel(const float* __restrict__ src,
                                                           unsigned short* __restrict__ dst,
                                                           int n) {
  int i = blockIdx.x * 256 + threadIdx.x;
  if (i < n) dst[i] = f2bf(src[i]);
}

// ============ GroupNorm stats: one block per (b,g) ============
__global__ __launch_bounds__(256) void gn_stats_kernel(const float* __restrict__ src,
                                                       float* __restrict__ stats) {
  const int bg = blockIdx.x;                  // group = 8 contiguous channels
  const float* base = src + (size_t)bg * 8 * NSP;
  float s = 0.f, sq = 0.f;
  for (int i = threadIdx.x; i < 8 * NSP; i += 256) {
    float v = base[i];
    s += v; sq += v * v;
  }
  __shared__ float sh1[256], sh2[256];
  sh1[threadIdx.x] = s; sh2[threadIdx.x] = sq;
  __syncthreads();
  for (int off = 128; off > 0; off >>= 1) {
    if (threadIdx.x < off) {
      sh1[threadIdx.x] += sh1[threadIdx.x + off];
      sh2[threadIdx.x] += sh2[threadIdx.x + off];
    }
    __syncthreads();
  }
  if (threadIdx.x == 0) {
    const float cnt = (float)(8 * NSP);
    float m = sh1[0] / cnt;
    stats[bg] = m;
    stats[64 + bg] = sh2[0] / cnt - m * m;
  }
}

// ============ GN apply + LDS-tiled transpose -> hT[b][n][c] bf16 ============
__global__ __launch_bounds__(256) void gn_apply_transpose_kernel(
    const float* __restrict__ x, const float* __restrict__ stats,
    const float* __restrict__ w, const float* __restrict__ bias,
    unsigned short* __restrict__ hT) {
  __shared__ float tile[32][33];
  const int n0 = blockIdx.x * 32, c0 = blockIdx.y * 32, b = blockIdx.z;
  const int j = threadIdx.x & 31, i0 = threadIdx.x >> 5;
  for (int i = i0; i < 32; i += 8) {
    const int c = c0 + i;
    const int sg = b * 32 + (c >> 3);
    float m = stats[sg], vv = stats[64 + sg];
    tile[i][j] = (x[((size_t)b * CCH + c) * NSP + n0 + j] - m) * rsqrtf(vv + EPSV) * w[c] + bias[c];
  }
  __syncthreads();
  for (int i = i0; i < 32; i += 8)
    hT[((size_t)b * NSP + n0 + i) * CCH + c0 + j] = f2bf(tile[j][i]);
}

// ============ pipelined WMMA GEMM over [o][c] x hT[n][c] ============
// IS_QKV=true : 768 rows; q -> qT[b][h][n][d] (elu+1), k (elu+1) & v -> kvbuf[b][0..511][n]
// IS_QKV=false: 256 rows; -> proj[b][o][n] f32
template <bool IS_QKV>
__global__ __launch_bounds__(256) void gemm_bf16_kernel(
    const unsigned short* __restrict__ wbf, const float* __restrict__ bias,
    const unsigned short* __restrict__ srcT,
    unsigned short* __restrict__ qTp, unsigned short* __restrict__ kvp,
    float* __restrict__ projp) {
  __shared__ unsigned short sbuf[2][16 * BPITCH];
  const int tid = threadIdx.x;
  const int lane = tid & 31;
  const int wave = tid >> 5;
  const int m0 = (blockIdx.x * 8 + wave) * 16;
  const int b = blockIdx.z;
  const int nbase = blockIdx.y * (16 * QKV_NT);
  const unsigned short* sTb = srcT + (size_t)b * NSP * CCH;

  TileU a[8];
#pragma unroll
  for (int kk = 0; kk < 8; ++kk) a[kk] = load_a_tile(wbf, CCH, m0, kk * 32, lane);

  const int row = tid >> 4;            // 0..15
  const int cb  = (tid & 15) * 16;     // 0..240, 32B per thread
  {
    const unsigned short* gp = sTb + (size_t)(nbase + row) * CCH + cb;
    unsigned short* lp = &sbuf[0][row * BPITCH + cb];
    async_copy16(gp,     lds_off32(lp));
    async_copy16(gp + 8, lds_off32(lp + 8));
  }

  for (int t = 0; t < QKV_NT; ++t) {
    wait_async0();
    __syncthreads();                    // tile t resident; prior consumers done
    if (t + 1 < QKV_NT) {
      const int n1 = nbase + (t + 1) * 16;
      const unsigned short* gp = sTb + (size_t)(n1 + row) * CCH + cb;
      unsigned short* lp = &sbuf[(t + 1) & 1][row * BPITCH + cb];
      async_copy16(gp,     lds_off32(lp));
      async_copy16(gp + 8, lds_off32(lp + 8));
    }
    const unsigned short* buf = sbuf[t & 1];
    f32x8 acc = {};
#pragma unroll
    for (int kk = 0; kk < 8; ++kk) {
      TileU bt = load_b_tile_lds(buf, kk * 32, lane);
      acc = wmma_bf16(a[kk], bt, acc);
    }
    const int n0 = nbase + t * 16;
    const int col = n0 + (lane & 15);
    const int rbase = (lane >> 4) * 8;
#pragma unroll
    for (int r = 0; r < 8; ++r) {
      const int o = m0 + rbase + r;
      float v = acc[r] + bias[o];
      if (IS_QKV) {
        float ex = __expf(fminf(v, 0.f));                 // branchless elu(x)+1
        float act = (v > 0.f) ? (v + 1.f) : ex;
        if (o < 256) {                                    // q -> per-head transposed layout
          qTp[((size_t)(b * NHEADS + (o >> 5)) * NSP + col) * HD + (o & 31)] = f2bf(act);
        } else {                                          // k (activated) and v (raw)
          float kv = (o < 512) ? act : v;
          kvp[((size_t)b * 512 + (o - 256)) * NSP + col] = f2bf(kv);
        }
      } else {
        projp[((size_t)b * CCH + o) * NSP + col] = v;
      }
    }
  }
}

// ============ kv[d][e] += k[d][n]*v[e][n]; ksum via ones-WMMA ============
#define KV_CHUNK 2048
__global__ __launch_bounds__(32) void kv_accum_kernel(
    const unsigned short* __restrict__ kvbuf,   // [B][512][N] bf16 (k rows 0..255, v rows 256..511)
    float* __restrict__ kv_raw,                 // [B][H][32][32]
    float* __restrict__ ksum) {                 // [B][H][32]
  const int lane = threadIdx.x;
  const int h = blockIdx.y, b = blockIdx.z;
  const unsigned short* kb = kvbuf + ((size_t)b * 512 + h * HD) * NSP;
  const unsigned short* vb = kvbuf + ((size_t)b * 512 + 256 + h * HD) * NSP;
  const int n0 = blockIdx.x * KV_CHUNK;

  TileU ones;                                   // B-tile of bf16 1.0: row-sum via matrix unit
#pragma unroll
  for (int i = 0; i < 16; ++i) ones.us[i] = 0x3F80;

  f32x8 acc00 = {}, acc01 = {}, acc10 = {}, acc11 = {};
  f32x8 ks0 = {}, ks1 = {};
  for (int n = n0; n < n0 + KV_CHUNK; n += 32) {
    __builtin_prefetch(kb + (size_t)(lane & 15) * NSP + n + 256, 0, 1);
    TileU a0 = load_a_tile(kb, NSP, 0, n, lane);
    TileU a1 = load_a_tile(kb, NSP, 16, n, lane);
    TileU b0 = load_b_tile(vb, NSP, 0, n, lane);
    TileU b1 = load_b_tile(vb, NSP, 16, n, lane);
    acc00 = wmma_bf16(a0, b0, acc00);
    acc01 = wmma_bf16(a0, b1, acc01);
    acc10 = wmma_bf16(a1, b0, acc10);
    acc11 = wmma_bf16(a1, b1, acc11);
    ks0   = wmma_bf16(a0, ones, ks0);           // D[m][*] = row-sum of k rows 0..15
    ks1   = wmma_bf16(a1, ones, ks1);           // rows 16..31
  }
  float* kvb = kv_raw + (size_t)(b * NHEADS + h) * HD * HD;
  const int col = lane & 15, rb = (lane >> 4) * 8;
#pragma unroll
  for (int r = 0; r < 8; ++r) {
    atomicAdd(&kvb[(0  + rb + r) * HD + 0  + col], acc00[r]);
    atomicAdd(&kvb[(0  + rb + r) * HD + 16 + col], acc01[r]);
    atomicAdd(&kvb[(16 + rb + r) * HD + 0  + col], acc10[r]);
    atomicAdd(&kvb[(16 + rb + r) * HD + 16 + col], acc11[r]);
  }
  if (col == 0) {                               // all columns equal; col 0 lanes commit
    float* ksb = ksum + (size_t)(b * NHEADS + h) * HD;
#pragma unroll
    for (int r = 0; r < 8; ++r) {
      atomicAdd(&ksb[rb + r],      ks0[r]);     // d = rb + r
      atomicAdd(&ksb[16 + rb + r], ks1[r]);     // d = 16 + rb + r
    }
  }
}

// ============ kvT[e][d] = kv[d][e]/clip(ksum[d]) as bf16; s[d] ============
__global__ __launch_bounds__(32) void kv_finalize_kernel(
    const float* __restrict__ kv_raw, const float* __restrict__ ksum,
    unsigned short* __restrict__ kvTbf, float* __restrict__ svec) {
  const int bh = blockIdx.x;
  const int d = threadIdx.x;
  float ks = ksum[bh * HD + d];
  float inv = 1.f / fmaxf(ks, 1e-6f);
  svec[bh * HD + d] = ks * inv;
  for (int e = 0; e < HD; ++e)
    kvTbf[(size_t)bh * HD * HD + e * HD + d] =
        f2bf(kv_raw[(size_t)bh * HD * HD + d * HD + e] * inv);
}

// ============ attention apply via WMMA: out[e][n] = (kvT x q) / clip(q.s) ============
__global__ __launch_bounds__(256) void attn_wmma_kernel(
    const unsigned short* __restrict__ qT,      // [B][H][N][32] bf16 (elu+1 applied)
    const unsigned short* __restrict__ kvTbf,   // [B*H][32(e)][32(d)] bf16
    const float* __restrict__ svec,             // [B*H][32]
    unsigned short* __restrict__ aT) {          // [B][N][256] bf16
  const int lane = threadIdx.x & 31;
  const int wave = threadIdx.x >> 5;
  const int bh = blockIdx.y;
  const int b = bh >> 3, h = bh & 7;
  const unsigned short* qTb = qT + (size_t)bh * NSP * HD;
  const unsigned short* kvb = kvTbf + (size_t)bh * HD * HD;

  TileU a0 = load_a_tile(kvb, HD, 0, 0, lane);    // e rows 0..15, K=d 0..31
  TileU a1 = load_a_tile(kvb, HD, 16, 0, lane);   // e rows 16..31
  float sv[16];
  const int kg = lane >> 4;
#pragma unroll
  for (int i = 0; i < 16; ++i) sv[i] = svec[bh * HD + kg * 16 + i];

  const int base = blockIdx.x * 2048 + wave * 256;
  for (int t = 0; t < 16; ++t) {
    const int n0 = base + t * 16;
    TileU bt = load_b_tile(qTb, HD, n0, 0, lane); // q^T tile: K=d contiguous per n
    float part = 0.f;
#pragma unroll
    for (int i = 0; i < 16; ++i) part += bf2f(bt.us[i]) * sv[i];
    part += __shfl_xor(part, 16, 32);             // combine the two K-halves
    const float inv = 1.f / fmaxf(part, 1e-6f);

    f32x8 acc0 = {}, acc1 = {};
    acc0 = wmma_bf16(a0, bt, acc0);
    acc1 = wmma_bf16(a1, bt, acc1);

    const int col = lane & 15;
    const int rbase = (lane >> 4) * 8;
    u16x8 p0, p1;
#pragma unroll
    for (int r = 0; r < 8; ++r) {
      p0[r] = f2bf(acc0[r] * inv);
      p1[r] = f2bf(acc1[r] * inv);
    }
    unsigned short* orow = aT + ((size_t)b * NSP + n0 + col) * CCH + h * HD;
    *(u16x8*)(orow + rbase)      = p0;            // e = rbase..rbase+7
    *(u16x8*)(orow + 16 + rbase) = p1;            // e = 16+rbase..
  }
}

// ============ GN2 + residual: out = x + gn(proj) ============
__global__ __launch_bounds__(256) void gn_norm_residual_kernel(
    const float* __restrict__ proj, const float* __restrict__ stats,
    const float* __restrict__ w, const float* __restrict__ bias,
    const float* __restrict__ x, float* __restrict__ out) {
  size_t idx = (size_t)blockIdx.x * 256 + threadIdx.x;
  int c = (int)((idx >> 15) & 255);
  int b = (int)(idx >> 23);
  int sg = b * 32 + (c >> 3);
  float m = stats[sg], v = stats[64 + sg];
  out[idx] = x[idx] + (proj[idx] - m) * rsqrtf(v + EPSV) * w[c] + bias[c];
}

// ======================= launcher =======================
extern "C" void kernel_launch(void* const* d_in, const int* in_sizes, int n_in,
                              void* d_out, int out_size, void* d_ws, size_t ws_size,
                              hipStream_t stream) {
  (void)in_sizes; (void)n_in; (void)out_size; (void)ws_size;
  const float* x      = (const float*)d_in[0];
  const float* norm_w = (const float*)d_in[1];
  const float* norm_b = (const float*)d_in[2];
  const float* qkv_w  = (const float*)d_in[3];
  const float* qkv_b  = (const float*)d_in[4];
  const float* out_w  = (const float*)d_in[5];
  const float* out_b  = (const float*)d_in[6];
  const float* onw    = (const float*)d_in[7];
  const float* onb    = (const float*)d_in[8];
  float* out = (float*)d_out;

  char* ws = (char*)d_ws;
  size_t off = 0;
  auto alloc = [&](size_t bytes) { size_t o = off; off += (bytes + 255) & ~(size_t)255; return o; };
  unsigned short* hT    = (unsigned short*)(ws + alloc((size_t)BB * NSP * CCH * 2));          // 32 MB
  unsigned short* qT    = (unsigned short*)(ws + alloc((size_t)BB * NHEADS * NSP * HD * 2));  // 32 MB
  unsigned short* kvbuf = (unsigned short*)(ws + alloc((size_t)BB * 512 * NSP * 2));          // 64 MB
  unsigned short* aT    = (unsigned short*)(ws + alloc((size_t)BB * NSP * CCH * 2));          // 32 MB
  float* proj           = (float*)(ws + alloc((size_t)BB * CCH * NSP * 4));                   // 64 MB
  unsigned short* wqbf  = (unsigned short*)(ws + alloc((size_t)O3 * CCH * 2));
  unsigned short* wobf  = (unsigned short*)(ws + alloc((size_t)CCH * CCH * 2));
  float* stats1         = (float*)(ws + alloc(128 * 4));
  float* stats2         = (float*)(ws + alloc(128 * 4));
  float* kv_raw         = (float*)(ws + alloc((size_t)BB * NHEADS * HD * HD * 4));
  float* ksum           = (float*)(ws + alloc((size_t)BB * NHEADS * HD * 4));
  unsigned short* kvTbf = (unsigned short*)(ws + alloc((size_t)BB * NHEADS * HD * HD * 2));
  float* svec           = (float*)(ws + alloc((size_t)BB * NHEADS * HD * 4));

  hipMemsetAsync(kv_raw, 0, (size_t)BB * NHEADS * HD * HD * 4, stream);
  hipMemsetAsync(ksum, 0, (size_t)BB * NHEADS * HD * 4, stream);

  f2bf_convert_kernel<<<(O3 * CCH) / 256, 256, 0, stream>>>(qkv_w, wqbf, O3 * CCH);
  f2bf_convert_kernel<<<(CCH * CCH) / 256, 256, 0, stream>>>(out_w, wobf, CCH * CCH);

  gn_stats_kernel<<<64, 256, 0, stream>>>(x, stats1);
  gn_apply_transpose_kernel<<<dim3(NSP / 32, CCH / 32, BB), 256, 0, stream>>>(x, stats1, norm_w, norm_b, hT);

  gemm_bf16_kernel<true><<<dim3(6, NSP / (16 * QKV_NT), BB), 256, 0, stream>>>(
      wqbf, qkv_b, hT, qT, kvbuf, nullptr);

  kv_accum_kernel<<<dim3(NSP / KV_CHUNK, NHEADS, BB), 32, 0, stream>>>(kvbuf, kv_raw, ksum);
  kv_finalize_kernel<<<BB * NHEADS, 32, 0, stream>>>(kv_raw, ksum, kvTbf, svec);
  attn_wmma_kernel<<<dim3(NSP / 2048, BB * NHEADS), 256, 0, stream>>>(qT, kvTbf, svec, aT);

  gemm_bf16_kernel<false><<<dim3(2, NSP / (16 * QKV_NT), BB), 256, 0, stream>>>(
      wobf, out_b, aT, nullptr, nullptr, proj);

  gn_stats_kernel<<<64, 256, 0, stream>>>(proj, stats2);
  gn_norm_residual_kernel<<<(BB * CCH * NSP) / 256, 256, 0, stream>>>(proj, stats2, onw, onb, x, out);
}
